// LSTMPriceModel_27590869909539
// MI455X (gfx1250) — 8-bit matrix operand order repaired and checked by reading, compiled, not run on hardware
//
#include <hip/hip_runtime.h>
#include <cmath>

#define BB   1024
#define SS   256
#define FF   5
#define TFF  7
#define HH   256
#define GG   1024
#define EPSF 1e-8f
#define WPITCH 272

typedef __attribute__((ext_vector_type(8)))  int   v8i;
typedef __attribute__((ext_vector_type(16))) int   v16i;
typedef __attribute__((ext_vector_type(8)))  float v8f;

__device__ __forceinline__ float sigmf(float x){ return 1.0f/(1.0f + __expf(-x)); }
__device__ __forceinline__ float clamp01(float v){ return fminf(fmaxf(v,0.f),1.f); }
__device__ __forceinline__ float cleanf(float v, float fill){
  return (fabsf(v) <= 3.0e38f) ? v : fill;
}

__device__ __forceinline__ unsigned char f32_fp8(float f){
  if (!(f==f)) return 0;
  unsigned s = (__float_as_uint(f)>>24)&0x80u;
  float af = fabsf(f);
  if (af < 9.765625e-4f) return (unsigned char)s;
  if (af > 448.f) af = 448.f;
  int e; float m = frexpf(af, &e);
  int E = e + 6;
  if (E >= 1) {
    int mant = (int)(m*16.f + 0.5f) - 8;
    if (mant >= 8){ mant = 0; E++; }
    if (E > 15 || (E==15 && mant>6)){ E=15; mant=6; }
    return (unsigned char)(s | (E<<3) | mant);
  } else {
    int mant = (int)(af*512.f + 0.5f);
    if (mant > 7) return (unsigned char)(s | (1<<3));
    return (unsigned char)(s | mant);
  }
}

__device__ __forceinline__ v16i load_a128(const unsigned char* p){
  v16i r;
  #pragma unroll
  for (int blk=0; blk<2; blk++){
    const unsigned char* q = p + blk*64;
    uint2 a = *(const uint2*)(q);
    uint2 b = *(const uint2*)(q+16);
    uint2 c = *(const uint2*)(q+32);
    uint2 d = *(const uint2*)(q+48);
    r[blk*8+0]=(int)a.x; r[blk*8+1]=(int)a.y; r[blk*8+2]=(int)b.x; r[blk*8+3]=(int)b.y;
    r[blk*8+4]=(int)c.x; r[blk*8+5]=(int)c.y; r[blk*8+6]=(int)d.x; r[blk*8+7]=(int)d.y;
  }
  return r;
}
__device__ __forceinline__ v16i load_b128k(const unsigned char* p){
  uint4 a = *(const uint4*)(p);
  uint4 b = *(const uint4*)(p+32);
  uint4 c = *(const uint4*)(p+64);
  uint4 d = *(const uint4*)(p+96);
  v16i r;
  r[0]=(int)a.x; r[1]=(int)a.y; r[2]=(int)a.z; r[3]=(int)a.w;
  r[4]=(int)b.x; r[5]=(int)b.y; r[6]=(int)b.z; r[7]=(int)b.w;
  r[8]=(int)c.x; r[9]=(int)c.y; r[10]=(int)c.z; r[11]=(int)c.w;
  r[12]=(int)d.x; r[13]=(int)d.y; r[14]=(int)d.z; r[15]=(int)d.w;
  return r;
}

__global__ void features_kernel(const float* __restrict__ x,
    const float* __restrict__ icen, const float* __restrict__ iscl,
    const float* __restrict__ tcen, const float* __restrict__ tscl,
    const float* __restrict__ pcen, const float* __restrict__ pscl,
    unsigned char* __restrict__ feats)
{
  int b = blockIdx.x*blockDim.x + threadIdx.x;
  if (b >= BB) return;
  const float pc0 = pcen[0], ps0 = pscl[0];
  float cb[20], c2b[20], gb[7], lb[7];
  #pragma unroll
  for (int i=0;i<20;i++){ cb[i]=0.f; c2b[i]=0.f; }
  #pragma unroll
  for (int i=0;i<7;i++){ gb[i]=0.f; lb[i]=0.f; }
  float s20=0.f, sq20=0.f, s7g=0.f, s7l=0.f;
  float prev=0.f, e10=0.f, e12=0.f, e26=0.f, em=0.f;
  for (int s=0;s<SS;s++){
    const float* xr = x + ((size_t)b*SS + s)*FF;
    float close = xr[3];
    float cn = clamp01((close - pc0)/(ps0 + EPSF));
    float pch, dlt;
    if (s==0){ pch=0.f; dlt=0.f; e10=cn; e12=cn; e26=cn; }
    else {
      pch = fminf(fmaxf(cn/(prev+EPSF)-1.f,-2.f),2.f);
      dlt = cn - prev;
      e10 = (2.f/11.f)*cn + (9.f/11.f)*e10;
      e12 = (2.f/13.f)*cn + (11.f/13.f)*e12;
      e26 = (2.f/27.f)*cn + (25.f/27.f)*e26;
    }
    float macd = e12 - e26;
    if (s==0) em = macd; else em = 0.2f*macd + 0.8f*em;
    float mh   = macd - em;
    float gain = fmaxf(dlt,0.f), loss = fmaxf(-dlt,0.f);
    float ag = (s>=7)? s7g*(1.f/7.f) : 0.f;
    float al = (s>=7)? s7l*(1.f/7.f) : 0.f;
    float rs  = fminf(fmaxf(ag/(al+EPSF),0.f),100.f);
    float rsi = cleanf(100.f - 100.f/(1.f+rs), 0.5f);
    float sma = (s>=20)? s20*(1.f/20.f) : 0.f;
    float m2  = (s>=20)? sq20 : 0.f;
    float var = (m2 - 20.f*sma*sma)*(1.f/19.f);
    float vol = sqrtf(fmaxf(var,0.f));
    float bbw = cleanf(4.f*vol/(sma+EPSF), 0.f);
    float tech[TFF] = {pch, e10, rsi, sma, mh, vol, bbw};
    unsigned char* fr = feats + ((size_t)b*SS + s)*16;
    #pragma unroll
    for (int i=0;i<FF;i++){
      float v = clamp01((xr[i]-icen[i])/(iscl[i]+EPSF));
      fr[i] = f32_fp8(v);
    }
    #pragma unroll
    for (int k=0;k<TFF;k++){
      float t = cleanf(tech[k], 0.f);
      float v = clamp01((t - tcen[k])/(tscl[k]+EPSF));
      fr[FF+k] = f32_fp8(v);
    }
    fr[12]=0; fr[13]=0; fr[14]=0; fr[15]=0;
    int i7 = s % 7, i20 = s % 20;
    s7g  += gain  - gb[i7];   gb[i7]  = gain;
    s7l  += loss  - lb[i7];   lb[i7]  = loss;
    s20  += cn    - cb[i20];  cb[i20] = cn;
    sq20 += cn*cn - c2b[i20]; c2b[i20]= cn*cn;
    prev = cn;
  }
}

__global__ void cvt_wih1_kernel(const float* __restrict__ w, unsigned char* __restrict__ o){
  int i = blockIdx.x*blockDim.x + threadIdx.x;
  if (i < GG*HH) o[i] = f32_fp8(w[i]);
}

template<int LAYER>
__global__ __launch_bounds__(256, 1)
void lstm_kernel(const float* __restrict__ Wih_f32,
                 const float* __restrict__ Whh_f32,
                 const float* __restrict__ bih,
                 const float* __restrict__ bhh,
                 const unsigned char* __restrict__ xin,
                 const unsigned char* __restrict__ wih_fp8,
                 unsigned char* __restrict__ hseq,
                 float* __restrict__ hlast)
{
  constexpr int WHH_B  = GG*WPITCH;
  constexpr int WIH_B  = (LAYER==0) ? GG*16 : 0;
  constexpr int HBUF_B = 16*WPITCH;
  __shared__ unsigned char sm[WHH_B + WIH_B + HBUF_B];
  unsigned char* whh  = sm;
  unsigned char* wih0 = sm + WHH_B;
  unsigned char* hbuf = sm + WHH_B + WIH_B;

  const int tid  = threadIdx.x;
  const int wv   = tid >> 5;
  const int lane = tid & 31;
  const int ln16 = lane & 15;
  const int hi   = lane >> 4;
  const int hi16 = hi*16;
  const int bt   = blockIdx.x;

  for (int n = tid*4; n < tid*4+4; n++){
    const float* wr = Whh_f32 + (size_t)n*HH;
    unsigned char* dr = whh + n*WPITCH;
    for (int k=0;k<HH;k++) dr[k] = f32_fp8(wr[k]);
    if (LAYER==0){
      const float* ir = Wih_f32 + (size_t)n*(FF+TFF);
      unsigned char* dd = wih0 + n*16;
      #pragma unroll
      for (int k=0;k<12;k++) dd[k] = f32_fp8(ir[k]);
      dd[12]=0; dd[13]=0; dd[14]=0; dd[15]=0;
    }
  }
  for (int i=tid;i<HBUF_B;i+=256) hbuf[i]=0;

  v8f biasC[4][2];
  const unsigned char* whhp[4][2];
  const unsigned char* wixp[4][2];
  #pragma unroll
  for (int g=0;g<4;g++)
    #pragma unroll
    for (int j=0;j<2;j++){
      const int n = g*HH + (2*wv+j)*16 + ln16;
      const float bv = bih[n] + bhh[n];
      #pragma unroll
      for (int r=0;r<8;r++) biasC[g][j][r] = bv;
        whhp[g][j] = whh + (size_t)n*WPITCH + hi*8;
      if (LAYER==0){
        wixp[g][j] = wih0 + (size_t)n*16 + hi*8;
      } else {
        wixp[g][j] = wih_fp8 + (size_t)n*HH + hi*8;
      }
    }

  v8f cst[2];
  #pragma unroll
  for (int j=0;j<2;j++)
    #pragma unroll
    for (int r=0;r<8;r++) cst[j][r]=0.f;

  __syncthreads();

  const int m    = ln16;
  const int gbt  = bt*16 + m;
  const int aoff = hi*8;
  const unsigned char* hrow = hbuf + m*WPITCH + aoff;
  const int crow = tid >> 4;
  const int ccol = (tid & 15) * 16;

  for (int s=0; s<SS; s++){
    v16i ax0 = {}, ax1 = {};
    v8i  axs = {};
    if (LAYER==0){
      const unsigned char* xp = xin + ((size_t)gbt*SS + s)*16 + aoff;
      uint2 t = *(const uint2*)xp;
      axs[0]=(int)t.x; axs[1]=(int)t.y;
    } else {
      const unsigned char* xp = xin + ((size_t)gbt*SS + s)*HH + aoff;
      ax0 = load_a128(xp);
      ax1 = load_a128(xp + 128);
    }
    v16i ah0 = load_a128(hrow);
    v16i ah1 = load_a128(hrow + 128);
    __syncthreads();

    v8f acc[4][2];
    #pragma unroll
    for (int g=0;g<4;g++)
      #pragma unroll
      for (int j=0;j<2;j++) acc[g][j] = biasC[g][j];

    #pragma unroll
    for (int kk=0; kk<2; kk++){
      #pragma unroll
      for (int g=0;g<4;g++)
        #pragma unroll
        for (int j=0;j<2;j++){
          v16i b = load_a128(whhp[g][j] + kk*128);
          acc[g][j] = __builtin_amdgcn_wmma_f32_16x16x128_fp8_fp8(
              kk ? ah1 : ah0, b, (short)0, acc[g][j], false, false);
        }
    }
    if (LAYER==0){
      #pragma unroll
      for (int g=0;g<4;g++)
        #pragma unroll
        for (int j=0;j<2;j++){
          uint2 lo = *(const uint2*)(wixp[g][j]);
          v8i bx = {};
          bx[0]=(int)lo.x; bx[1]=(int)lo.y;
          acc[g][j] = __builtin_amdgcn_wmma_f32_16x16x64_fp8_fp8(
              axs, bx, (short)0, acc[g][j], false, false);
        }
    } else {
      #pragma unroll
      for (int kk=0; kk<2; kk++){
        #pragma unroll
        for (int g=0;g<4;g++)
          #pragma unroll
          for (int j=0;j<2;j++){
            v16i b = load_a128(wixp[g][j] + kk*128);
            acc[g][j] = __builtin_amdgcn_wmma_f32_16x16x128_fp8_fp8(
                kk ? ax1 : ax0, b, (short)0, acc[g][j], false, false);
          }
      }
    }

    #pragma unroll
    for (int j=0;j<2;j++){
      const int col = (2*wv+j)*16 + ln16;
      #pragma unroll
      for (int r=0;r<8;r++){
        float c = sigmf(acc[1][j][r])*cst[j][r] + sigmf(acc[0][j][r])*tanhf(acc[2][j][r]);
        cst[j][r] = c;
        float h = sigmf(acc[3][j][r])*tanhf(c);
        const int row = r + hi*8;
        hbuf[row*WPITCH + col] = f32_fp8(h);
        if (LAYER==1 && s==SS-1) hlast[(size_t)(bt*16+row)*HH + col] = h;
      }
    }
    __syncthreads();

    if (LAYER==0){
      uint4 v = *(const uint4*)(hbuf + crow*WPITCH + ccol);
      *(uint4*)(hseq + ((size_t)(bt*16+crow)*SS + s)*HH + ccol) = v;
    }
  }
}

__global__ void head_kernel(const float* __restrict__ hl, const float* __restrict__ fw,
                            const float* __restrict__ fb, const float* __restrict__ pcen,
                            const float* __restrict__ pscl, float* __restrict__ out){
  int b = blockIdx.x*blockDim.x + threadIdx.x;
  if (b >= BB) return;
  float acc = fb[0];
  const float* hr = hl + (size_t)b*HH;
  for (int k=0;k<HH;k++) acc += hr[k]*fw[k];
  out[b] = sigmf(acc)*pscl[0] + pcen[0];
}

extern "C" void kernel_launch(void* const* d_in, const int* in_sizes, int n_in,
                              void* d_out, int out_size, void* d_ws, size_t ws_size,
                              hipStream_t stream){
  const float* x    = (const float*)d_in[0];
  const float* icen = (const float*)d_in[1];
  const float* iscl = (const float*)d_in[2];
  const float* tcen = (const float*)d_in[3];
  const float* tscl = (const float*)d_in[4];
  const float* pcen = (const float*)d_in[5];
  const float* pscl = (const float*)d_in[6];
  const float* Wih0 = (const float*)d_in[7];
  const float* Whh0 = (const float*)d_in[8];
  const float* bih0 = (const float*)d_in[9];
  const float* bhh0 = (const float*)d_in[10];
  const float* Wih1 = (const float*)d_in[11];
  const float* Whh1 = (const float*)d_in[12];
  const float* bih1 = (const float*)d_in[13];
  const float* bhh1 = (const float*)d_in[14];
  const float* fcw  = (const float*)d_in[15];
  const float* fcb  = (const float*)d_in[16];
  float* out = (float*)d_out;

  unsigned char* ws    = (unsigned char*)d_ws;
  unsigned char* feats = ws;
  unsigned char* h1    = ws    + (size_t)BB*SS*16;
  unsigned char* wih1q = h1    + (size_t)BB*SS*HH;
  float*         hlast = (float*)(wih1q + (size_t)GG*HH);

  cvt_wih1_kernel <<<dim3((GG*HH+255)/256), dim3(256), 0, stream>>>(Wih1, wih1q);
  features_kernel <<<dim3(BB/256), dim3(256), 0, stream>>>(x,icen,iscl,tcen,tscl,pcen,pscl,feats);
  lstm_kernel<0>  <<<dim3(BB/16), dim3(256), 0, stream>>>(Wih0,Whh0,bih0,bhh0,feats,nullptr,h1,nullptr);
  lstm_kernel<1>  <<<dim3(BB/16), dim3(256), 0, stream>>>(Wih1,Whh1,bih1,bhh1,h1,wih1q,nullptr,hlast);
  head_kernel     <<<dim3(BB/256), dim3(256), 0, stream>>>(hlast,fcw,fcb,pcen,pscl,out);
}
